// MHA_51032801411558
// MI455X (gfx1250) — compile-verified, matrix operand order checked by reading, not run on hardware
//
#include <hip/hip_runtime.h>
#include <hip/hip_bf16.h>
#include <cstdint>

// Problem constants (match reference)
#define BATCH 2
#define SEQ   2048
#define DIM   1024
#define NH    16
#define DKH   64
#define TD    (3*DIM)   // 3072

typedef __attribute__((ext_vector_type(16))) __bf16        v16bf;
typedef __attribute__((ext_vector_type(8)))  float         v8f;
typedef __attribute__((ext_vector_type(4)))  unsigned int  u32x4;
typedef int v4i_b __attribute__((vector_size(16)));   // matches builtin's V4i

union FragBF { v16bf f; u32x4 u[2]; };

#if defined(__gfx1250__) && __has_builtin(__builtin_amdgcn_global_load_async_to_lds_b128)
#define HAVE_ASYNC_LDS 1
#else
#define HAVE_ASYNC_LDS 0
#endif

__device__ __forceinline__ unsigned short f2bf(float x) {
  unsigned u = __float_as_uint(x);
  u += 0x7FFFu + ((u >> 16) & 1u);          // round-to-nearest-even
  return (unsigned short)(u >> 16);
}

__device__ __forceinline__ v8f v8f_zero() {
  v8f z = {0.f,0.f,0.f,0.f,0.f,0.f,0.f,0.f};
  return z;
}

// ---------------------------------------------------------------- fp32 -> bf16
__global__ void cvt_f32_bf16(const float* __restrict__ s,
                             unsigned short* __restrict__ d, int n) {
  int i = blockIdx.x * blockDim.x + threadIdx.x;
  if (i < n) d[i] = f2bf(s[i]);
}

// --------------------------------------------- fp32 W[k][n] -> bf16 Wt[n][k]
// 32x32 tiled transpose through LDS; coalesced on both sides.
__global__ __launch_bounds__(256)
void cvtT_f32_bf16(const float* __restrict__ W, unsigned short* __restrict__ Wt,
                   int K, int N) {
  __shared__ float sh[32 * 33];
  const int nk = K / 32;
  const int k0 = (blockIdx.x % nk) * 32;
  const int n0 = (blockIdx.x / nk) * 32;
  const int tid = threadIdx.x;
  #pragma unroll
  for (int p = 0; p < 4; p++) {
    int e = tid + p * 256, r = e >> 5, c = e & 31;
    sh[r * 33 + c] = W[(size_t)(k0 + r) * N + n0 + c];
  }
  __syncthreads();
  #pragma unroll
  for (int p = 0; p < 4; p++) {
    int e = tid + p * 256, r = e >> 5, c = e & 31;   // r = n, c = k
    Wt[(size_t)(n0 + r) * K + k0 + c] = f2bf(sh[c * 33 + r]);
  }
}

// ---------------------------------------------------------------- mask -> bits
// One bit per mask element via wave32 ballot: 32x less DRAM per head-pass.
__global__ void pack_mask(const int* __restrict__ m, unsigned* __restrict__ bits) {
  int i = blockIdx.x * blockDim.x + threadIdx.x;
  unsigned b = (unsigned)__ballot(m[i] != 0);
  if ((threadIdx.x & 31) == 0) bits[i >> 5] = b;
}

// ------------------------------------------- V slice of qkv -> vT[b][feat][l]
// 64x64 tiled u16 transpose so attention sees contiguous per-feature rows.
__global__ __launch_bounds__(256)
void transpose_v(const unsigned short* __restrict__ qkv,
                 unsigned short* __restrict__ vtb) {
  __shared__ unsigned short sh[64 * 72];   // [feat][l]
  const int nt_l = SEQ / 64, nt_f = DIM / 64;
  const int ft = blockIdx.x % nt_f;
  const int lt = (blockIdx.x / nt_f) % nt_l;
  const int b  = blockIdx.x / (nt_f * nt_l);
  const int l0 = lt * 64, f0 = ft * 64;
  const int tid = threadIdx.x;
  #pragma unroll
  for (int p = 0; p < 8; p++) {
    int e = tid + p * 256, l = e >> 5, fp = e & 31;
    unsigned w = *(const unsigned*)(
        qkv + (size_t)(b * SEQ + l0 + l) * TD + 2 * DIM + f0 + 2 * fp);
    sh[(2 * fp    ) * 72 + l] = (unsigned short)(w & 0xFFFFu);
    sh[(2 * fp + 1) * 72 + l] = (unsigned short)(w >> 16);
  }
  __syncthreads();
  #pragma unroll
  for (int p = 0; p < 8; p++) {
    int e = tid + p * 256, f = e >> 5, lp = e & 31;
    unsigned w = (unsigned)sh[f * 72 + 2 * lp] |
                 ((unsigned)sh[f * 72 + 2 * lp + 1] << 16);
    *(unsigned*)(vtb + ((size_t)b * DIM + f0 + f) * SEQ + l0 + 2 * lp) = w;
  }
}

// ---------------------------------------------------------------- bf16 GEMM
// C[M,N] = A[M,K] * Bt[N,K]^T + bias ; both operands row-contiguous in K, so
// LDS staging is pure b128 copies. 128x128 block tile, 8 waves, 32x64/wave.
template<bool OUT_BF16>
__global__ __launch_bounds__(256)
void gemm_bf16(const unsigned short* __restrict__ A,
               const unsigned short* __restrict__ Bt,
               const float* __restrict__ bias,
               void* __restrict__ Cout,
               int M, int N, int K)
{
  __shared__ __align__(16) unsigned short As[128 * 48];   // [row][k], pitch 48
  __shared__ __align__(16) unsigned short Bs[128 * 48];   // [n][k],  pitch 48

  const int tid  = threadIdx.x;
  const int lane = tid & 31;
  const int wave = tid >> 5;
  const int lm   = lane & 15;
  const int hb   = (lane & 16) ? 8 : 0;     // per-lane K-base / C row half
  const int wm   = (wave & 3) * 32;
  const int wn   = (wave >> 2) * 64;
  const int M0   = blockIdx.y * 128;
  const int N0   = blockIdx.x * 128;

  v8f acc[2][4];
  #pragma unroll
  for (int i = 0; i < 2; i++)
    #pragma unroll
    for (int j = 0; j < 4; j++) acc[i][j] = v8f_zero();

  for (int k0 = 0; k0 < K; k0 += 32) {
    #pragma unroll
    for (int p = 0; p < 2; p++) {
      int idx = tid + p * 256;
      int row = idx >> 2, q = idx & 3;
      *(u32x4*)&As[row * 48 + q * 8] =
          *(const u32x4*)(A + (size_t)(M0 + row) * K + k0 + q * 8);
      *(u32x4*)&Bs[row * 48 + q * 8] =
          *(const u32x4*)(Bt + (size_t)(N0 + row) * K + k0 + q * 8);
    }
    __syncthreads();

    FragBF fa[2], fb;
    #pragma unroll
    for (int i = 0; i < 2; i++) {
      const unsigned short* p = &As[(wm + i * 16 + lm) * 48 + hb];
      fa[i].u[0] = *(const u32x4*)p;
      fa[i].u[1] = *(const u32x4*)(p + 16);
    }
    #pragma unroll
    for (int j = 0; j < 4; j++) {
      const unsigned short* p = &Bs[(wn + j * 16 + lm) * 48 + hb];
      fb.u[0] = *(const u32x4*)p;
      fb.u[1] = *(const u32x4*)(p + 16);
      #pragma unroll
      for (int i = 0; i < 2; i++)
        acc[i][j] = __builtin_amdgcn_wmma_f32_16x16x32_bf16(
            false, fa[i].f, false, fb.f, (short)0, acc[i][j], false, false);
    }
    __syncthreads();
  }

  #pragma unroll
  for (int i = 0; i < 2; i++) {
    #pragma unroll
    for (int j = 0; j < 4; j++) {
      int col = N0 + wn + j * 16 + lm;
      float bv = bias[col];
      #pragma unroll
      for (int e = 0; e < 8; e++) {
        int row = M0 + wm + i * 16 + hb + e;
        float v = acc[i][j][e] + bv;
        if (OUT_BF16) ((unsigned short*)Cout)[(size_t)row * N + col] = f2bf(v);
        else          ((float*)Cout)[(size_t)row * N + col] = v;
      }
    }
  }
}

// ---------------------------------------------------------------- flash attention
// One block = 128 query rows of one (b,h); 8 waves, 16 q-rows each.
// Q/K fragments straight from global; V tile async-copied to LDS from the
// pre-transposed vT buffer; mask read as packed bits.
__global__ __launch_bounds__(256)
void attn_flash(const unsigned short* __restrict__ qkv,   // [B*L][3D] bf16
                const unsigned short* __restrict__ vtb,   // [B][D][L] bf16
                const unsigned* __restrict__ mbits,       // [B][L][L/32]
                unsigned short* __restrict__ obuf)        // [B*L][D]  bf16
{
  __shared__ __align__(16) unsigned short Vt[64 * 72];       // [feature][key]
  __shared__ __align__(16) unsigned short Pst[8][16 * 72];   // per-wave P staging

  const int tid  = threadIdx.x;
  const int lane = tid & 31;
  const int wave = tid >> 5;
  const int lm   = lane & 15;
  const int hb   = (lane & 16) ? 8 : 0;

  const int nq = SEQ / 128;
  const int qt = blockIdx.x % nq;
  const int h  = (blockIdx.x / nq) % NH;
  const int b  = blockIdx.x / (nq * NH);

  const int q0 = qt * 128 + wave * 16;

  // Q fragments (rows q0+lm, 64 features)
  FragBF fq[2];
  #pragma unroll
  for (int kk = 0; kk < 2; kk++) {
    const unsigned short* p =
        qkv + (size_t)(b * SEQ + q0 + lm) * TD + h * DKH + kk * 32 + hb;
    fq[kk].u[0] = *(const u32x4*)p;
    fq[kk].u[1] = *(const u32x4*)(p + 16);
  }

  float mrun[8], lrun[8];
  v8f O[4];
  #pragma unroll
  for (int e = 0; e < 8; e++) { mrun[e] = -1e30f; lrun[e] = 0.0f; }
  #pragma unroll
  for (int n = 0; n < 4; n++) O[n] = v8f_zero();

  for (int kc = 0; kc < SEQ; kc += 64) {
    __syncthreads();
    // ---- V tile: 64 contiguous feature-rows of 64 keys -> LDS [feat][key]
    {
      #pragma unroll
      for (int p = 0; p < 2; p++) {
        int e = tid + p * 256;
        int f = e >> 3, g = e & 7;
        const unsigned short* src =
            vtb + ((size_t)b * DIM + h * DKH + f) * SEQ + kc + g * 8;
        unsigned short* dst = &Vt[f * 72 + g * 8];
#if HAVE_ASYNC_LDS
        __builtin_amdgcn_global_load_async_to_lds_b128(
            (__attribute__((address_space(1))) v4i_b*)src,
            (__attribute__((address_space(3))) v4i_b*)dst, 0, 0);
#else
        *(u32x4*)dst = *(const u32x4*)src;
#endif
      }
#if HAVE_ASYNC_LDS
#if __has_builtin(__builtin_amdgcn_s_wait_asynccnt)
      __builtin_amdgcn_s_wait_asynccnt(0);
#else
      asm volatile("s_wait_asynccnt 0x0" ::: "memory");
#endif
#endif
    }
    __syncthreads();

    // ---- packed mask words (uniform per 16-lane half)
    unsigned mw0[8], mw1[8];
    #pragma unroll
    for (int e = 0; e < 8; e++) {
      size_t base = (size_t)(b * SEQ + q0 + hb + e) * (SEQ / 32) + (kc >> 5);
      mw0[e] = mbits[base];
      mw1[e] = mbits[base + 1];
    }

    // ---- S = QK^T * scale with masked_fill-to-0 semantics
    v8f S[4];
    float pm[4][8];
    #pragma unroll
    for (int n = 0; n < 4; n++) {
      FragBF fk0, fk1;
      const unsigned short* p =
          qkv + (size_t)(b * SEQ + kc + n * 16 + lm) * TD + DIM + h * DKH + hb;
      fk0.u[0] = *(const u32x4*)p;        fk0.u[1] = *(const u32x4*)(p + 16);
      fk1.u[0] = *(const u32x4*)(p + 32); fk1.u[1] = *(const u32x4*)(p + 48);
      v8f s = v8f_zero();
      s = __builtin_amdgcn_wmma_f32_16x16x32_bf16(false, fq[0].f, false, fk0.f,
                                                  (short)0, s, false, false);
      s = __builtin_amdgcn_wmma_f32_16x16x32_bf16(false, fq[1].f, false, fk1.f,
                                                  (short)0, s, false, false);
      const int bit = ((n & 1) << 4) + lm;
      #pragma unroll
      for (int e = 0; e < 8; e++) {
        unsigned w = (n < 2) ? mw0[e] : mw1[e];
        float pmv = ((w >> bit) & 1u) ? 1.0f : 0.0f;
        pm[n][e] = pmv;
        s[e] = s[e] * 0.125f * pmv;
      }
      S[n] = s;
    }

    // ---- online softmax stats (denominator keeps masked exp(0-m) terms)
    float mnew[8], alpha[8], rsum[8];
    #pragma unroll
    for (int e = 0; e < 8; e++) {
      float v = fmaxf(fmaxf(S[0][e], S[1][e]), fmaxf(S[2][e], S[3][e]));
      v = fmaxf(v, __shfl_xor(v, 1, 32));
      v = fmaxf(v, __shfl_xor(v, 2, 32));
      v = fmaxf(v, __shfl_xor(v, 4, 32));
      v = fmaxf(v, __shfl_xor(v, 8, 32));
      mnew[e]  = fmaxf(mrun[e], v);
      alpha[e] = __expf(mrun[e] - mnew[e]);
      mrun[e]  = mnew[e];
      rsum[e]  = 0.0f;
    }
    #pragma unroll
    for (int n = 0; n < 4; n++) {
      #pragma unroll
      for (int e = 0; e < 8; e++) {
        float pfull = __expf(S[n][e] - mnew[e]);
        rsum[e] += pfull;
        Pst[wave][(hb + e) * 72 + n * 16 + lm] = f2bf(pfull * pm[n][e]);
      }
    }
    #pragma unroll
    for (int e = 0; e < 8; e++) {
      float r = rsum[e];
      r += __shfl_xor(r, 1, 32);
      r += __shfl_xor(r, 2, 32);
      r += __shfl_xor(r, 4, 32);
      r += __shfl_xor(r, 8, 32);
      lrun[e] = lrun[e] * alpha[e] + r;
      #pragma unroll
      for (int n = 0; n < 4; n++) O[n][e] *= alpha[e];
    }

    // ---- P fragments from per-wave staging (same-wave LDS ordering)
    FragBF fp[2];
    #pragma unroll
    for (int kk = 0; kk < 2; kk++) {
      const unsigned short* p = &Pst[wave][lm * 72 + kk * 32 + hb];
      fp[kk].u[0] = *(const u32x4*)p;
      fp[kk].u[1] = *(const u32x4*)(p + 16);
    }
    // ---- O += P * V
    #pragma unroll
    for (int n = 0; n < 4; n++) {
      FragBF fv0, fv1;
      const unsigned short* p = &Vt[(n * 16 + lm) * 72 + hb];
      fv0.u[0] = *(const u32x4*)p;        fv0.u[1] = *(const u32x4*)(p + 16);
      fv1.u[0] = *(const u32x4*)(p + 32); fv1.u[1] = *(const u32x4*)(p + 48);
      O[n] = __builtin_amdgcn_wmma_f32_16x16x32_bf16(false, fp[0].f, false, fv0.f,
                                                     (short)0, O[n], false, false);
      O[n] = __builtin_amdgcn_wmma_f32_16x16x32_bf16(false, fp[1].f, false, fv1.f,
                                                     (short)0, O[n], false, false);
    }
  }

  // ---- normalize + write merged-head bf16 output
  #pragma unroll
  for (int n = 0; n < 4; n++) {
    #pragma unroll
    for (int e = 0; e < 8; e++) {
      float v = O[n][e] / lrun[e];
      obuf[(size_t)(b * SEQ + q0 + hb + e) * DIM + h * DKH + n * 16 + lm] = f2bf(v);
    }
  }
}

// ---------------------------------------------------------------- launcher
extern "C" void kernel_launch(void* const* d_in, const int* in_sizes, int n_in,
                              void* d_out, int out_size, void* d_ws, size_t ws_size,
                              hipStream_t stream)
{
  const float* x    = (const float*)d_in[0];
  const int*   mask = (const int*)  d_in[1];
  const float* Wqkv = (const float*)d_in[2];
  const float* bqkv = (const float*)d_in[3];
  const float* Wout = (const float*)d_in[4];
  const float* bout = (const float*)d_in[5];
  float* out = (float*)d_out;

  // workspace layout (bytes), total ~58 MB
  char* ws = (char*)d_ws;
  unsigned short* xb     = (unsigned short*)(ws);             //  8 MB  x bf16
  unsigned short* wqkvT  = (unsigned short*)(ws +  8388608);  //  6 MB  W_qkv^T bf16 [3072][1024]
  unsigned short* woutT  = (unsigned short*)(ws + 14680064);  //  2 MB  W_out^T bf16 [1024][1024]
  unsigned short* qkvb   = (unsigned short*)(ws + 16777216);  // 24 MB  qkv bf16
  unsigned short* ob     = (unsigned short*)(ws + 41943040);  //  8 MB  attn out bf16
  unsigned short* vtb    = (unsigned short*)(ws + 50331648);  //  8 MB  V^T bf16 [B][D][L]
  unsigned*       mbits  = (unsigned*)      (ws + 58720256);  //  1 MB  packed mask

  const int nx = BATCH * SEQ * DIM;                 // 4194304
  cvt_f32_bf16<<<nx / 256, 256, 0, stream>>>(x, xb, nx);
  cvtT_f32_bf16<<<(DIM / 32) * (TD / 32),  256, 0, stream>>>(Wqkv, wqkvT, DIM, TD);
  cvtT_f32_bf16<<<(DIM / 32) * (DIM / 32), 256, 0, stream>>>(Wout, woutT, DIM, DIM);
  pack_mask<<<(BATCH * SEQ * SEQ) / 256, 256, 0, stream>>>(mask, mbits);

  // qkv = x @ W_qkv + b_qkv   (M=4096, N=3072, K=1024), bf16 out
  gemm_bf16<true><<<dim3(TD / 128, (BATCH * SEQ) / 128), 256, 0, stream>>>(
      xb, wqkvT, bqkv, qkvb, BATCH * SEQ, TD, DIM);

  // V slice -> [b][feat][l]
  transpose_v<<<BATCH * (SEQ / 64) * (DIM / 64), 256, 0, stream>>>(qkvb, vtb);

  // flash attention per (b, h, 128-row q tile)
  attn_flash<<<BATCH * NH * (SEQ / 128), 256, 0, stream>>>(qkvb, vtb, mbits, ob);

  // out = attn_out @ W_out + b_out  (M=4096, N=1024, K=1024), fp32 out
  gemm_bf16<false><<<dim3(DIM / 128, (BATCH * SEQ) / 128), 256, 0, stream>>>(
      ob, woutT, bout, out, BATCH * SEQ, DIM, DIM);
}
